// SpatialDGCNN_20100446945953
// MI455X (gfx1250) — compile-verified, matrix operand order checked by reading, not run on hardware
//
#include <hip/hip_runtime.h>
#include <math.h>

// ---------------------------------------------------------------------------
// CDNA5 (gfx1250) SpatialDGCNN forward pass.
//   * All GEMM-shaped stages use v_wmma_f32_16x16x32_f16 (f16 LDS tiles,
//     f32 accumulation). Fragments are loaded as 2x ds_load_b128 per operand
//     (row-major [16][32] tiles; B kept transposed in LDS), staged from
//     global with float4 (global_load_b128) + 4x4 register transposes +
//     ds_store_b64 writes.
//   * KNN distance GEMM is fused with a streaming top-17 selector so the
//     4x2048x2048 distance matrix never touches HBM (workload is memory
//     bound: ~100 GFLOP vs ~1-2 GB traffic at 23.3 TB/s).
//   * BN statistics stream channel rows into LDS via the Tensor Data Mover
//     (tensor_load_to_lds + s_wait_tensorcnt) — async-tensor path.
// ---------------------------------------------------------------------------

typedef __attribute__((ext_vector_type(16))) _Float16 v16h;
typedef __attribute__((ext_vector_type(8)))  _Float16 v8h;
typedef __attribute__((ext_vector_type(4)))  _Float16 v4h;
typedef __attribute__((ext_vector_type(8)))  float    v8f;
typedef __attribute__((ext_vector_type(4)))  unsigned v4u;
typedef __attribute__((ext_vector_type(8)))  unsigned v8u;

namespace {
constexpr int Bn  = 4;            // batch
constexpr int Np  = 2048;         // points
constexpr int KNB = 16;           // neighbors
constexpr int NKt = Np * KNB;     // 32768 columns per batch
constexpr int NCc = 8;            // num classes
constexpr float LNEG = 0.2f;
constexpr float EPS  = 1e-5f;
}

// ---- WMMA fragment loader --------------------------------------------------
// T points at a row-major [16][ld] f16 tile (ld in halfs, 16B aligned rows).
// ISA 7.12.2: lane's 8 VGPRs cover k = {hi*8..hi*8+7} and {16+hi*8..+7},
// i.e. exactly two contiguous 16-byte runs -> two ds_load_b128.
__device__ __forceinline__ v16h frag_row(const _Float16* T, int ld) {
  int lane = threadIdx.x & 31;
  int r = lane & 15, hi = lane >> 4;
  v8h lo = *(const v8h*)&T[r * ld + hi * 8];
  v8h hh = *(const v8h*)&T[r * ld + 16 + hi * 8];
  v16h out;
#pragma unroll
  for (int i = 0; i < 8; ++i) { out[i] = lo[i]; out[8 + i] = hh[i]; }
  return out;
}

__device__ __forceinline__ v8f wmma_f16(v16h a, v16h b, v8f c) {
  return __builtin_amdgcn_wmma_f32_16x16x32_f16(false, a, false, b, (short)0,
                                                c, false, false);
}

// ---- TDM: 1-row tensor_load_to_lds (D# groups 0/1 per ISA 8.3/8.4) --------
// Copies nElems f32 (nElems <= 65535) from 'src' to LDS offset 'ldsOff'.
// Wave-level instruction: call from one wave, then s_wait_tensorcnt + barrier.
__device__ __forceinline__ void tdm_load_row(const float* src, unsigned ldsOff,
                                             unsigned nElems) {
  unsigned long long ga = (unsigned long long)(size_t)src;
  v4u g0;
  g0[0] = 1u;                                         // count=1, user mode
  g0[1] = ldsOff;                                     // lds_addr
  g0[2] = (unsigned)(ga & 0xffffffffu);               // global_addr[31:0]
  g0[3] = (unsigned)((ga >> 32) & 0x01ffffffu)        // global_addr[56:32]
          | (2u << 30);                               // type = 2 ("image")
  v8u g1;
  g1[0] = 2u << 16;                                   // data_size = 4 bytes
  g1[1] = (nElems & 0xffffu) << 16;                   // tensor_dim0 lo16
  g1[2] = (nElems >> 16) | (1u << 16);                // dim0 hi16 | dim1 lo16=1
  g1[3] = (nElems << 16);                             // dim1 hi=0 | tile_dim0
  g1[4] = 1u;                                         // tile_dim1=1, tile_dim2=0
  g1[5] = nElems;                                     // tensor_dim0_stride lo32
  g1[6] = 0u;                                         // stride0 hi | stride1 lo
  g1[7] = 0u;
  asm volatile("tensor_load_to_lds %0, %1" :: "s"(g0), "s"(g1) : "memory");
}

// ---- generic WMMA GEMM: C[b] = op(A[b]) * B[b] ----------------------------
// Block = 128 threads (4 waves); block tile 32(M) x 64(N); each wave owns a
// 32x16 column strip -> 2 wmma per k-chunk with a shared B fragment.
__global__ void gemm_wmma_kernel(const float* __restrict__ A,
                                 const float* __restrict__ Bm,
                                 float* __restrict__ C,
                                 int M, int Ncols, int Kdim,
                                 long strideA, long strideB, long strideC,
                                 int lda, int ldb, int ldc, int transA) {
  __shared__ __align__(16) _Float16 As[32][32];
  __shared__ __align__(16) _Float16 Bs[4][16][32];   // transposed: [n][k]
  const int wave = threadIdx.x >> 5;
  const int lane = threadIdx.x & 31;
  const int b = blockIdx.z;
  const float* Ab = A + (long)b * strideA;
  const float* Bb = Bm + (long)b * strideB;
  float* Cb = C + (long)b * strideC;
  const int mBase = blockIdx.y * 32;
  const int nBase = (blockIdx.x * 4 + wave) * 16;
  v8f acc0 = {}, acc1 = {};
  for (int k0 = 0; k0 < Kdim; k0 += 32) {
    // ---- stage A 32x32: 256 float4 quads, 2 per thread ----
    if (!transA) {
      for (int q = threadIdx.x; q < 256; q += blockDim.x) {
        int m = q >> 3, kq = (q & 7) * 4;
        int gm = mBase + m, gk = k0 + kq;
        v4h h = {};
        if (gm < M) {
          if (gk + 3 < Kdim) {
            const float4 f = *(const float4*)&Ab[(long)gm * lda + gk];
            h[0] = (_Float16)f.x; h[1] = (_Float16)f.y;
            h[2] = (_Float16)f.z; h[3] = (_Float16)f.w;
          } else {
#pragma unroll
            for (int j = 0; j < 4; ++j)
              h[j] = (_Float16)((gk + j < Kdim) ? Ab[(long)gm * lda + gk + j]
                                                : 0.f);
          }
        }
        *(v4h*)&As[m][kq] = h;
      }
    } else {  // small F x F attention weights: scalar path is fine
      for (int t = threadIdx.x; t < 32 * 32; t += blockDim.x) {
        int m = t >> 5, kk = t & 31;
        int gm = mBase + m, gk = k0 + kk;
        As[m][kk] = (_Float16)((gm < M && gk < Kdim)
                                   ? Ab[(long)gk * lda + gm] : 0.f);
      }
    }
    // ---- stage B 32x16 per wave: 4x4 blocks, transpose into [n][k] ----
    {
      int kb = (lane >> 2) * 4, nb = (lane & 3) * 4;
      int gn = nBase + nb;
      float r[4][4];
#pragma unroll
      for (int kk = 0; kk < 4; ++kk) {
        int gk = k0 + kb + kk;
        if (gk < Kdim && gn + 3 < Ncols) {
          const float4 f = *(const float4*)&Bb[(long)gk * ldb + gn];
          r[kk][0] = f.x; r[kk][1] = f.y; r[kk][2] = f.z; r[kk][3] = f.w;
        } else {
#pragma unroll
          for (int j = 0; j < 4; ++j)
            r[kk][j] = (gk < Kdim && gn + j < Ncols)
                           ? Bb[(long)gk * ldb + gn + j] : 0.f;
        }
      }
#pragma unroll
      for (int nn = 0; nn < 4; ++nn) {
        v4h h = {(_Float16)r[0][nn], (_Float16)r[1][nn],
                 (_Float16)r[2][nn], (_Float16)r[3][nn]};
        *(v4h*)&Bs[wave][nb + nn][kb] = h;
      }
    }
    if (k0 + 32 < Kdim)   // pull next B slab toward GL2
      __builtin_prefetch(&Bb[(long)(k0 + 32) * ldb + nBase + lane], 0, 1);
    __syncthreads();
    {
      v16h bf = frag_row(&Bs[wave][0][0], 32);
      v16h a0 = frag_row(&As[0][0], 32);
      v16h a1 = frag_row(&As[16][0], 32);
      acc0 = wmma_f16(a0, bf, acc0);
      acc1 = wmma_f16(a1, bf, acc1);
    }
    __syncthreads();
  }
  // D layout: lane n = lane&15; VGPR v -> row v (+8 for lanes 16-31)
  const int n = lane & 15;
  const int mo = (lane >> 4) * 8;
#pragma unroll
  for (int v = 0; v < 8; ++v) {
    int gn = nBase + n;
    int g0m = mBase + mo + v, g1m = mBase + 16 + mo + v;
    if (g0m < M && gn < Ncols) Cb[(long)g0m * ldc + gn] = acc0[v];
    if (g1m < M && gn < Ncols) Cb[(long)g1m * ldc + gn] = acc1[v];
  }
}

// ---- squared norms of the KNN source channels -----------------------------
__global__ void norms_kernel(const float* __restrict__ feat, int C,
                             long batchStride, float* __restrict__ norms) {
  int n = blockIdx.x * blockDim.x + threadIdx.x;
  int b = blockIdx.z;
  if (n >= Np) return;
  const float* fb = feat + (long)b * batchStride;
  float s = 0.f;
  for (int c = 0; c < C; ++c) { float v = fb[(long)c * Np + n]; s += v * v; }
  norms[b * Np + n] = s;
}

// ---- fused KNN: WMMA inner products + streaming top-17 --------------------
__global__ void knn_topk_kernel(const float* __restrict__ feat, int C,
                                long batchStride,
                                const float* __restrict__ norms,
                                int* __restrict__ idx) {
  __shared__ __align__(16) _Float16 As[16][256];
  __shared__ __align__(16) _Float16 Bs[16][32];      // transposed: [n][k]
  __shared__ float Sc[16][16];
  const int lane = threadIdx.x;
  const int b = blockIdx.z;
  const int rowBase = blockIdx.x * 16;
  const float* fb = feat + (long)b * batchStride;
  const float* nrm = norms + b * Np;
  const int Cp = (C + 31) & ~31;            // zero-pad K to WMMA chunks
  // preload A rows (16 x Cp) via 4x4 blocks: feat contiguous along points
  for (int t = lane; t < Cp; t += 32) {
    int m0 = (t & 3) * 4, c0 = (t >> 2) * 4;
    float r[4][4];
#pragma unroll
    for (int cc = 0; cc < 4; ++cc) {
      int c = c0 + cc;
      if (c < C) {
        const float4 f = *(const float4*)&fb[(long)c * Np + rowBase + m0];
        r[cc][0] = f.x; r[cc][1] = f.y; r[cc][2] = f.z; r[cc][3] = f.w;
      } else {
        r[cc][0] = r[cc][1] = r[cc][2] = r[cc][3] = 0.f;
      }
    }
#pragma unroll
    for (int mm = 0; mm < 4; ++mm) {
      v4h h = {(_Float16)r[0][mm], (_Float16)r[1][mm],
               (_Float16)r[2][mm], (_Float16)r[3][mm]};
      *(v4h*)&As[m0 + mm][c0] = h;
    }
  }
  float best[KNB + 1];
  int   besti[KNB + 1];
#pragma unroll
  for (int j = 0; j <= KNB; ++j) { best[j] = -1e30f; besti[j] = 0; }
  __syncthreads();

  for (int colBase = 0; colBase < Np; colBase += 16) {
    v8f acc = {};
    for (int k0 = 0; k0 < Cp; k0 += 32) {
      {  // stage B 32x16 -> transposed [n][k], one 4x4 block per lane
        int cb = (lane >> 2) * 4, nb = (lane & 3) * 4;
        float r[4][4];
#pragma unroll
        for (int cc = 0; cc < 4; ++cc) {
          int c = k0 + cb + cc;
          if (c < C) {
            const float4 f = *(const float4*)&fb[(long)c * Np + colBase + nb];
            r[cc][0] = f.x; r[cc][1] = f.y; r[cc][2] = f.z; r[cc][3] = f.w;
          } else {
            r[cc][0] = r[cc][1] = r[cc][2] = r[cc][3] = 0.f;
          }
        }
#pragma unroll
        for (int nn = 0; nn < 4; ++nn) {
          v4h h = {(_Float16)r[0][nn], (_Float16)r[1][nn],
                   (_Float16)r[2][nn], (_Float16)r[3][nn]};
          *(v4h*)&Bs[nb + nn][cb] = h;
        }
      }
      __syncthreads();
      v16h af = frag_row(&As[0][k0], 256);
      v16h bf = frag_row(&Bs[0][0], 32);
      acc = wmma_f16(af, bf, acc);
      __syncthreads();
    }
    // neg_dist = 2*inner - xx_row - xx_col
    const int n = lane & 15;
    const int mo = (lane >> 4) * 8;
#pragma unroll
    for (int v = 0; v < 8; ++v)
      Sc[mo + v][n] = 2.f * acc[v] - nrm[rowBase + mo + v] - nrm[colBase + n];
    __syncthreads();
    if (lane < 16) {                        // row 'lane': merge 16 candidates
      for (int n2 = 0; n2 < 16; ++n2) {
        float s = Sc[lane][n2];
        if (s > best[KNB]) {
          int j = KNB;
          while (j > 0 && best[j - 1] < s) {
            best[j] = best[j - 1]; besti[j] = besti[j - 1]; --j;
          }
          best[j] = s; besti[j] = colBase + n2;
        }
      }
    }
    __syncthreads();
  }
  if (lane < 16) {                          // drop slot 0 (self match)
    long base = ((long)b * Np + rowBase + lane) * KNB;
    for (int j = 0; j < KNB; ++j) idx[base + j] = besti[j + 1];
  }
}

// ---- graph feature: gf[c2, n*16+k] = (c2<C ? neigh-center : center) -------
__global__ void build_gf_kernel(const float* __restrict__ feat, int C,
                                long batchStride, const int* __restrict__ idx,
                                float* __restrict__ gf) {
  long t = (long)blockIdx.x * blockDim.x + threadIdx.x;
  int b = blockIdx.z;
  long total = (long)2 * C * NKt;
  if (t >= total) return;
  int col = (int)(t % NKt);
  int c2  = (int)(t / NKt);
  int n = col >> 4, kk = col & 15;
  int c = (c2 < C) ? c2 : c2 - C;
  const float* fb = feat + (long)b * batchStride;
  float center = fb[(long)c * Np + n];
  float v;
  if (c2 < C) {
    int j = idx[((long)b * Np + n) * KNB + kk];
    v = fb[(long)c * Np + j] - center;
  } else {
    v = center;
  }
  gf[(long)b * 2 * C * NKt + (long)c2 * NKt + col] = v;
}

// ---- BN statistics: TDM-stream channel rows into LDS, reduce --------------
__global__ void bn_stats_kernel(const float* __restrict__ y, long count, int nb,
                                long batchStride, float* __restrict__ mean,
                                float* __restrict__ istd) {
  constexpr int CH = 2048;                  // count is a multiple of 2048
  __shared__ __align__(16) float buf[CH];
  __shared__ float s1[256], s2[256];
  int o = blockIdx.x;
  float a1 = 0.f, a2 = 0.f;
  for (int b = 0; b < nb; ++b) {
    const float* row = y + (long)b * batchStride + (long)o * count;
    for (long c0 = 0; c0 < count; c0 += CH) {
      if (threadIdx.x < 32) {               // wave 0 drives the TDM
        tdm_load_row(row + c0, (unsigned)(size_t)buf, CH);
        __builtin_amdgcn_s_wait_tensorcnt(0);
      }
      __syncthreads();
      for (int i = threadIdx.x; i < CH; i += blockDim.x) {
        float v = buf[i];
        a1 += v; a2 += v * v;
      }
      __syncthreads();
    }
  }
  s1[threadIdx.x] = a1; s2[threadIdx.x] = a2;
  __syncthreads();
  for (int s = 128; s > 0; s >>= 1) {
    if (threadIdx.x < (unsigned)s) {
      s1[threadIdx.x] += s1[threadIdx.x + s];
      s2[threadIdx.x] += s2[threadIdx.x + s];
    }
    __syncthreads();
  }
  if (threadIdx.x == 0) {
    long tot = count * nb;
    float m = s1[0] / (float)tot;
    float var = s2[0] / (float)tot - m * m;
    mean[o] = m;
    istd[o] = rsqrtf(var + EPS);
  }
}

__global__ void bn_apply_kernel(float* __restrict__ y, long count,
                                long batchStride, const float* __restrict__ mean,
                                const float* __restrict__ istd,
                                const float* __restrict__ g,
                                const float* __restrict__ bb, int lrelu) {
  int o = blockIdx.y, b = blockIdx.z;
  long col = (long)blockIdx.x * blockDim.x + threadIdx.x;
  if (col >= count) return;
  long i = (long)b * batchStride + (long)o * count + col;
  float v = (y[i] - mean[o]) * istd[o] * g[o] + bb[o];
  if (lrelu) v = v >= 0.f ? v : LNEG * v;
  y[i] = v;
}

// ---- fused attention tail: coeff lookup, lrelu, softmax(k), elu, mean(k) --
__global__ void att_fuse_kernel(const float* __restrict__ Wh, int F,
                                const float* __restrict__ x,
                                const int* __restrict__ idx,
                                const float* __restrict__ Apair,
                                float* __restrict__ xout, long outBatchStride,
                                int chanOfs) {
  int n = blockIdx.x * blockDim.x + threadIdx.x;
  int g = blockIdx.y, b = blockIdx.z;
  if (n >= Np) return;
  int c = (int)x[(long)b * 3 * Np + 2 * Np + n];
  const float* whb = Wh + (long)b * F * NKt + (long)g * NKt + (long)n * KNB;
  float e[KNB], wv[KNB];
  float emax = -1e30f;
#pragma unroll
  for (int kk = 0; kk < KNB; ++kk) {
    int j = idx[((long)b * Np + n) * KNB + kk];
    int tcl = (int)x[(long)b * 3 * Np + 2 * Np + j];
    int lo = c < tcl ? c : tcl;
    int hi = c < tcl ? tcl : c;
    int pidx = lo * NCc - (lo * (lo - 1)) / 2 + (hi - lo);
    float a = Apair[(long)pidx * F + g];
    float w = whb[kk];
    float ee = w * a;
    ee = ee >= 0.f ? ee : LNEG * ee;
    e[kk] = ee; wv[kk] = w;
    emax = fmaxf(emax, ee);
  }
  float den = 0.f;
#pragma unroll
  for (int kk = 0; kk < KNB; ++kk) { e[kk] = expf(e[kk] - emax); den += e[kk]; }
  float accm = 0.f;
#pragma unroll
  for (int kk = 0; kk < KNB; ++kk) {
    float hp = (e[kk] / den) * wv[kk];
    hp = hp > 0.f ? hp : expf(hp) - 1.f;     // elu
    accm += hp;
  }
  xout[(long)b * outBatchStride + (long)(chanOfs + g) * Np + n] =
      accm * (1.f / KNB);
}

// ---- max+mean pooling over N ----------------------------------------------
__global__ void pool_kernel(const float* __restrict__ x5,
                            float* __restrict__ pooled) {
  int c = blockIdx.x, b = blockIdx.z;
  __shared__ float sm[256], ss[256];
  const float* p = x5 + (long)b * 1024 * Np + (long)c * Np;
  float mx = -1e30f, sum = 0.f;
  for (int i = threadIdx.x; i < Np; i += blockDim.x) {
    float v = p[i];
    mx = fmaxf(mx, v);
    sum += v;
  }
  sm[threadIdx.x] = mx; ss[threadIdx.x] = sum;
  __syncthreads();
  for (int s = 128; s > 0; s >>= 1) {
    if (threadIdx.x < (unsigned)s) {
      sm[threadIdx.x] = fmaxf(sm[threadIdx.x], sm[threadIdx.x + s]);
      ss[threadIdx.x] += ss[threadIdx.x + s];
    }
    __syncthreads();
  }
  if (threadIdx.x == 0) {
    pooled[b * 2048 + c] = sm[0];
    pooled[b * 2048 + 1024 + c] = ss[0] / (float)Np;
  }
}

// ---- small dense layers (B=4: too skinny for WMMA, dot-reduce) ------------
__global__ void linear_kernel(const float* __restrict__ in,
                              const float* __restrict__ W,
                              const float* __restrict__ bias,
                              float* __restrict__ out, int IN, int OUT) {
  int o = blockIdx.x, b = blockIdx.z;
  __shared__ float s[256];
  float a = 0.f;
  const float* ib = in + (long)b * IN;
  const float* wr = W + (long)o * IN;
  for (int i = threadIdx.x; i < IN; i += blockDim.x) a += ib[i] * wr[i];
  s[threadIdx.x] = a;
  __syncthreads();
  for (int st = 128; st > 0; st >>= 1) {
    if (threadIdx.x < (unsigned)st) s[threadIdx.x] += s[threadIdx.x + st];
    __syncthreads();
  }
  if (threadIdx.x == 0) out[b * OUT + o] = s[0] + (bias ? bias[o] : 0.f);
}

__global__ void bn_rows_kernel(float* __restrict__ y, int F, int nb,
                               const float* __restrict__ g,
                               const float* __restrict__ bb, int lrelu) {
  int f = blockIdx.x * blockDim.x + threadIdx.x;
  if (f >= F) return;
  float m = 0.f;
  for (int b = 0; b < nb; ++b) m += y[b * F + f];
  m /= nb;
  float var = 0.f;
  for (int b = 0; b < nb; ++b) { float d = y[b * F + f] - m; var += d * d; }
  var /= nb;
  float is = rsqrtf(var + EPS);
  for (int b = 0; b < nb; ++b) {
    float v = (y[b * F + f] - m) * is * g[f] + bb[f];
    if (lrelu) v = v >= 0.f ? v : LNEG * v;
    y[b * F + f] = v;
  }
}

// ---------------------------------------------------------------------------
extern "C" void kernel_launch(void* const* d_in, const int* in_sizes, int n_in,
                              void* d_out, int out_size, void* d_ws,
                              size_t ws_size, hipStream_t stream) {
  (void)in_sizes; (void)n_in; (void)out_size; (void)ws_size;
  const float* x = (const float*)d_in[0];
  const float* convw[4] = {(const float*)d_in[1], (const float*)d_in[6],
                           (const float*)d_in[11], (const float*)d_in[16]};
  const float* bng[4] = {(const float*)d_in[2], (const float*)d_in[7],
                         (const float*)d_in[12], (const float*)d_in[17]};
  const float* bnb[4] = {(const float*)d_in[3], (const float*)d_in[8],
                         (const float*)d_in[13], (const float*)d_in[18]};
  const float* attW[4] = {(const float*)d_in[4], (const float*)d_in[9],
                          (const float*)d_in[14], (const float*)d_in[19]};
  const float* attA[4] = {(const float*)d_in[5], (const float*)d_in[10],
                          (const float*)d_in[15], (const float*)d_in[20]};
  const float* conv5w = (const float*)d_in[21];
  const float* bn5g = (const float*)d_in[22];
  const float* bn5b = (const float*)d_in[23];
  const float* lin1w = (const float*)d_in[24];
  const float* bn6g = (const float*)d_in[25];
  const float* bn6b = (const float*)d_in[26];
  const float* lin2w = (const float*)d_in[27];
  const float* lin2b = (const float*)d_in[28];
  const float* bn7g = (const float*)d_in[29];
  const float* bn7b = (const float*)d_in[30];
  const float* lin3w = (const float*)d_in[31];
  const float* lin3b = (const float*)d_in[32];

  char* wp = (char*)d_ws;
  auto bump = [&](size_t bytes) -> void* {
    void* p = (void*)wp;
    wp += (bytes + 255) & ~(size_t)255;
    return p;
  };
  float* norms  = (float*)bump(sizeof(float) * Bn * Np);
  int*   idx    = (int*)bump(sizeof(int) * (size_t)Bn * Np * KNB);
  float* xc     = (float*)bump(sizeof(float) * (size_t)Bn * 1024 * Np);
  float* gf     = (float*)bump(sizeof(float) * (size_t)Bn * 512 * NKt);
  float* hbuf   = (float*)bump(sizeof(float) * (size_t)Bn * 256 * NKt);
  float* whbuf  = (float*)bump(sizeof(float) * (size_t)Bn * 256 * NKt);
  float* y5     = (float*)bump(sizeof(float) * (size_t)Bn * 1024 * Np);
  float* meanb  = (float*)bump(sizeof(float) * 1024);
  float* istdb  = (float*)bump(sizeof(float) * 1024);
  float* pooled = (float*)bump(sizeof(float) * Bn * 2048);
  float* h6     = (float*)bump(sizeof(float) * Bn * 512);
  float* h7     = (float*)bump(sizeof(float) * Bn * 256);

  for (int L = 0; L < 4; ++L) {
    const float* feat; int Cfeat; long fbs; int knnC; int F; int chanBase;
    if (L == 0)      { feat = x;             Cfeat = 3;   fbs = 3L * Np;    knnC = 2;   F = 64;  chanBase = 0;   }
    else if (L == 1) { feat = xc;            Cfeat = 128; fbs = 1024L * Np; knnC = 128; F = 64;  chanBase = 128; }
    else if (L == 2) { feat = xc + 128 * Np; Cfeat = 128; fbs = 1024L * Np; knnC = 128; F = 128; chanBase = 256; }
    else             { feat = xc + 256 * Np; Cfeat = 256; fbs = 1024L * Np; knnC = 256; F = 256; chanBase = 512; }

    norms_kernel<<<dim3((Np + 255) / 256, 1, Bn), 256, 0, stream>>>(
        feat, knnC, fbs, norms);
    knn_topk_kernel<<<dim3(Np / 16, 1, Bn), 32, 0, stream>>>(
        feat, knnC, fbs, norms, idx);

    long tot = 2L * Cfeat * NKt;
    build_gf_kernel<<<dim3((unsigned)((tot + 255) / 256), 1, Bn), 256, 0, stream>>>(
        feat, Cfeat, fbs, idx, gf);

    const int O = F, Kd = 2 * Cfeat;
    gemm_wmma_kernel<<<dim3(NKt / 64, O / 32, Bn), 128, 0, stream>>>(
        convw[L], gf, hbuf, O, NKt, Kd,
        0L, (long)2 * Cfeat * NKt, (long)F * NKt, Kd, NKt, NKt, 0);
    bn_stats_kernel<<<O, 256, 0, stream>>>(hbuf, (long)NKt, Bn, (long)F * NKt,
                                           meanb, istdb);
    bn_apply_kernel<<<dim3(NKt / 256, O, Bn), 256, 0, stream>>>(
        hbuf, (long)NKt, (long)F * NKt, meanb, istdb, bng[L], bnb[L], 1);

    for (int hd = 0; hd < 2; ++hd) {
      gemm_wmma_kernel<<<dim3(NKt / 64, F / 32, Bn), 128, 0, stream>>>(
          attW[L] + (size_t)hd * F * F, hbuf, whbuf, F, NKt, F,
          0L, (long)F * NKt, (long)F * NKt, F, NKt, NKt, 1);
      att_fuse_kernel<<<dim3((Np + 127) / 128, F, Bn), 128, 0, stream>>>(
          whbuf, F, x, idx, attA[L] + (size_t)hd * 36 * F, xc, 1024L * Np,
          chanBase + hd * F);
    }
  }

  // conv5 (1024x1024) over xc, BN + lrelu, pool, MLP head
  gemm_wmma_kernel<<<dim3(Np / 64, 1024 / 32, Bn), 128, 0, stream>>>(
      conv5w, xc, y5, 1024, Np, 1024,
      0L, 1024L * Np, 1024L * Np, 1024, Np, Np, 0);
  bn_stats_kernel<<<1024, 256, 0, stream>>>(y5, (long)Np, Bn, 1024L * Np,
                                            meanb, istdb);
  bn_apply_kernel<<<dim3(Np / 256, 1024, Bn), 256, 0, stream>>>(
      y5, (long)Np, 1024L * Np, meanb, istdb, bn5g, bn5b, 1);
  pool_kernel<<<dim3(1024, 1, Bn), 256, 0, stream>>>(y5, pooled);

  linear_kernel<<<dim3(512, 1, Bn), 256, 0, stream>>>(pooled, lin1w, nullptr,
                                                      h6, 2048, 512);
  bn_rows_kernel<<<(512 + 255) / 256, 256, 0, stream>>>(h6, 512, Bn, bn6g,
                                                        bn6b, 1);
  linear_kernel<<<dim3(256, 1, Bn), 256, 0, stream>>>(h6, lin2w, lin2b, h7,
                                                      512, 256);
  bn_rows_kernel<<<(256 + 255) / 256, 256, 0, stream>>>(h7, 256, Bn, bn7g,
                                                        bn7b, 1);
  linear_kernel<<<dim3(16, 1, Bn), 256, 0, stream>>>(h7, lin3w, lin3b,
                                                     (float*)d_out, 256, 16);
}
